// DisNCELoss_78967268704379
// MI455X (gfx1250) — compile-verified, matrix operand order checked by reading, not run on hardware
//
#include <hip/hip_runtime.h>
#include <hip/hip_bf16.h>

#define POSN 512
#define NEGN 512
#define NTOT 1024
#define DIMK 128
#define NBATCH 64
#define INVT (1.0f / 0.07f)
// exp((t-1)*INVT) == exp2(t*EK1 + EK2)
#define EK1 (INVT * 1.4426950408889634f)
#define EK2 (-(INVT * 1.4426950408889634f))

typedef __attribute__((ext_vector_type(16))) __bf16 v16bf;
typedef __attribute__((ext_vector_type(8)))  float  v8f;

union V16U {
  uint4    q[2];
  unsigned u[8];
  v16bf    v;
};

// round-to-nearest-even f32 pair -> packed bf16 dword
__device__ __forceinline__ unsigned pk_bf16(float lo, float hi) {
  unsigned a = __float_as_uint(lo), b = __float_as_uint(hi);
  a += 0x7fffu + ((a >> 16) & 1u);
  b += 0x7fffu + ((b >> 16) & 1u);
  return (a >> 16) | (b & 0xffff0000u);
}

__device__ __forceinline__ const float* row_ptr(const float* __restrict__ fB,
                                                const float* __restrict__ fR,
                                                int b, int n) {
  return (n < POSN) ? (fB + (size_t)(b * POSN + n) * DIMK)
                    : (fR + (size_t)(b * NEGN + (n - POSN)) * DIMK);
}

__global__ void zero_kernel(float* out, int n) {
  int i = blockIdx.x * blockDim.x + threadIdx.x;
  if (i < n) out[i] = 0.0f;
}

__global__ __launch_bounds__(256) void disnce_kernel(const float* __restrict__ featB,
                                                     const float* __restrict__ featR,
                                                     float* __restrict__ out) {
  // 128 rows x 128 bf16, row stride padded 128->136 shorts (272B) to break
  // the stride-256B (exact 64-bank) conflict pattern on ds_load_b128.
  __shared__ __align__(16) unsigned short lds[128 * 136];

  const int b      = blockIdx.x >> 2;          // batch
  const int wgRow0 = (blockIdx.x & 3) * 256;   // 256 rows per workgroup
  const int tid    = threadIdx.x;
  const int lane   = tid & 31;
  const int wave   = tid >> 5;                 // 8 waves
  const int lHalf  = (lane >> 4) & 1;
  const int l15    = lane & 15;

  // ================= analytic positive-sum prologue (pure f32) ===========
  // sum_{n} sum_{m in class(n), m!=n} dot(f_n,f_m)
  //   = |sum_{m in P} f_m|^2 + |sum_{m in N} f_m|^2 - sum_n ||f_n||^2
  // Each WG owns 256 rows of one class; 2 WGs/class each take half of |csum|^2.
  const int classBase = (wgRow0 < POSN) ? 0 : POSN;
  {
    // class-sum vector: wave w sums 64 of the 512 class rows, lane = k-slice
    float cx = 0.f, cy = 0.f, cz = 0.f, cw = 0.f;
    const int k4 = lane * 4;
    for (int m = 0; m < 64; ++m) {
      const float* src = row_ptr(featB, featR, b, classBase + wave * 64 + m) + k4;
      float4 f = *(const float4*)src;
      cx += f.x; cy += f.y; cz += f.z; cw += f.w;
    }
    float* fscr = (float*)lds;  // 8 waves x 32 lanes x float4 = 4KB scratch
    float4 st; st.x = cx; st.y = cy; st.z = cz; st.w = cw;
    *(float4*)&fscr[(wave * 32 + lane) * 4] = st;
    __syncthreads();

    float qpart = 0.f;
    if (tid < 32) {
      float sx = 0.f, sy = 0.f, sz = 0.f, sw = 0.f;
      for (int w = 0; w < 8; ++w) {
        float4 t = *(const float4*)&fscr[(w * 32 + tid) * 4];
        sx += t.x; sy += t.y; sz += t.z; sw += t.w;
      }
      qpart = sx * sx + sy * sy + sz * sz + sw * sw;
#pragma unroll
      for (int m = 1; m < 32; m <<= 1) qpart += __shfl_xor(qpart, m, 32);
    }

    // per-row self-norms (this WG's 256 rows), reduced per wave
    float nrm = 0.f;
    const float* src = row_ptr(featB, featR, b, wgRow0 + tid);
    for (int k = 0; k < DIMK; k += 4) {
      float4 f = *(const float4*)(src + k);
      nrm += f.x * f.x + f.y * f.y + f.z * f.z + f.w * f.w;
    }
    float val = nrm * (INVT / 511.0f / 1024.0f);  // +(1/511T) * ||f_n||^2
#pragma unroll
    for (int m = 1; m < 32; m <<= 1) val += __shfl_xor(val, m, 32);
    if (lane == 0) atomicAdd(out, val);
    if (tid == 0) {
      // constant C per row (256 rows) and -(1/511T) * |csum|^2 / 2
      atomicAdd(out, 256.0f * INVT / 1024.0f -
                         qpart * (0.5f * INVT / 511.0f / 1024.0f));
    }
  }
  // (first pass's barrier below protects fscr before LDS reuse)

  // ================= A operands: 2 strips x (16 rows x 128 K) ============
  // 16-bit A 16x32 layout: lanes 0-15 = rows, K {0..7,16..23}; lanes 16-31
  // same rows, K {8..15,24..31}. Loaded once from global f32, converted.
  V16U a[2][4];
#pragma unroll
  for (int s = 0; s < 2; ++s) {
    const int r0 = wgRow0 + (wave * 2 + s) * 16;
    const float* src = row_ptr(featB, featR, b, r0 + l15);
#pragma unroll
    for (int c = 0; c < 4; ++c) {
      const int kb = c * 32 + lHalf * 8;
      float4 f0 = *(const float4*)(src + kb);
      float4 f1 = *(const float4*)(src + kb + 4);
      float4 f2 = *(const float4*)(src + kb + 16);
      float4 f3 = *(const float4*)(src + kb + 20);
      a[s][c].u[0] = pk_bf16(f0.x, f0.y);
      a[s][c].u[1] = pk_bf16(f0.z, f0.w);
      a[s][c].u[2] = pk_bf16(f1.x, f1.y);
      a[s][c].u[3] = pk_bf16(f1.z, f1.w);
      a[s][c].u[4] = pk_bf16(f2.x, f2.y);
      a[s][c].u[5] = pk_bf16(f2.z, f2.w);
      a[s][c].u[6] = pk_bf16(f3.x, f3.y);
      a[s][c].u[7] = pk_bf16(f3.z, f3.w);
    }
  }

  // E = sum_{m} exp((dot-1)/T)  (self INCLUDED here; cancelled exactly below)
  // C/D layout: lane (l15) = column, VGPR j = row (lower/upper half = +0/+8)
  float E[2][8];
#pragma unroll
  for (int s = 0; s < 2; ++s)
#pragma unroll
    for (int j = 0; j < 8; ++j) E[s][j] = 0.0f;

  // ========= stream columns in 8 blocks of 128 via LDS (branch-free) =====
  for (int p = 0; p < 8; ++p) {
    const int blockStart = p * 128;
    __syncthreads();  // previous block's B reads (or prologue scratch) done
    for (int i = tid; i < 128 * 32; i += 256) {
      const int row = i >> 5;
      const int k4  = (i & 31) * 4;
      const float* src = row_ptr(featB, featR, b, blockStart + row) + k4;
      float4 f = *(const float4*)src;
      uint2 st;
      st.x = pk_bf16(f.x, f.y);
      st.y = pk_bf16(f.z, f.w);
      *(uint2*)(&lds[row * 136 + k4]) = st;
    }
    __syncthreads();

#pragma unroll
    for (int s = 0; s < 2; ++s) {
      for (int ct = 0; ct < 8; ++ct) {
        // B operand tile (Gram symmetry: identical lane pattern to A)
        V16U bt[4];
        const unsigned short* bp = &lds[(ct * 16 + l15) * 136 + lHalf * 8];
#pragma unroll
        for (int c = 0; c < 4; ++c) {
          bt[c].q[0] = *(const uint4*)(bp + c * 32);
          bt[c].q[1] = *(const uint4*)(bp + c * 32 + 16);
        }
        v8f acc = {0.f, 0.f, 0.f, 0.f, 0.f, 0.f, 0.f, 0.f};
#pragma unroll
        for (int c = 0; c < 4; ++c) {
          acc = __builtin_amdgcn_wmma_f32_16x16x32_bf16(
              false, a[s][c].v, false, bt[c].v, (short)0, acc, false, false);
        }
#pragma unroll
        for (int j = 0; j < 8; ++j)
          E[s][j] += __builtin_amdgcn_exp2f(fmaf(acc[j], EK1, EK2));
      }
    }
  }

  // ===== exact self-term cancellation: recompute the diagonal tile ========
  // B operand for the strip's own 16 rows == the A registers (Gram symmetry),
  // and they hold bit-identical bf16 data to what the LDS path fed the main
  // loop, so this exp is bit-identical to the one accumulated above.
#pragma unroll
  for (int s = 0; s < 2; ++s) {
    v8f acc = {0.f, 0.f, 0.f, 0.f, 0.f, 0.f, 0.f, 0.f};
#pragma unroll
    for (int c = 0; c < 4; ++c) {
      acc = __builtin_amdgcn_wmma_f32_16x16x32_bf16(
          false, a[s][c].v, false, a[s][c].v, (short)0, acc, false, false);
    }
#pragma unroll
    for (int j = 0; j < 8; ++j) {
      const float selfE = __builtin_amdgcn_exp2f(fmaf(acc[j], EK1, EK2));
      // diagonal element of row (j + 8*lHalf) sits in column-lane l15 == that
      E[s][j] -= (l15 == j + lHalf * 8) ? selfE : 0.0f;
    }
  }

  // ============ per-row log-sum-exp: reduce 16 column-lanes per half =====
  float local = 0.0f;
#pragma unroll
  for (int s = 0; s < 2; ++s) {
#pragma unroll
    for (int j = 0; j < 8; ++j) {
      float e = E[s][j];
#pragma unroll
      for (int m = 1; m < 16; m <<= 1) e += __shfl_xor(e, m, 32);
      local += __logf(e);
    }
  }
  // lanes 0-15 replicate the lower-half 8-row sum, 16-31 the upper-half sum
  local += __shfl_xor(local, 16, 32);
  if (lane == 0) atomicAdd(out, local * (1.0f / 1024.0f));
}

extern "C" void kernel_launch(void* const* d_in, const int* in_sizes, int n_in,
                              void* d_out, int out_size, void* d_ws, size_t ws_size,
                              hipStream_t stream) {
  (void)in_sizes; (void)n_in; (void)d_ws; (void)ws_size;
  const float* featB = (const float*)d_in[0];
  const float* featR = (const float*)d_in[1];
  float* out = (float*)d_out;
  zero_kernel<<<1, 64, 0, stream>>>(out, out_size);
  disnce_kernel<<<NBATCH * 4, 256, 0, stream>>>(featB, featR, out);
}
